// HeavyCompressor_70643622085010
// MI455X (gfx1250) — matrix operand order repaired and checked by reading, not compiled, not run on hardware
//
#include <hip/hip_runtime.h>
#include <hip/hip_bf16.h>
#include <stdint.h>

typedef __attribute__((ext_vector_type(16))) _Float16 v16h;
typedef __attribute__((ext_vector_type(8)))  _Float16 v8h;
typedef __attribute__((ext_vector_type(4)))  _Float16 v4h;
typedef __attribute__((ext_vector_type(8)))  float    v8f;
typedef __attribute__((ext_vector_type(4)))  float    v4f;
typedef __attribute__((ext_vector_type(4)))  int      v4i;

#define B_   4
#define S_   8192
#define D_   2048
#define HD_  128
#define M_   8
#define N_   (S_ / M_)
#define KCHUNK 128
#define ROWT 4                  // 16-row tiles per block
#define ROWS (ROWT * 16)        // 64 S-rows per block

// gfx1250 async global->LDS path, guarded so the build never breaks.
#if __has_builtin(__builtin_amdgcn_global_load_async_to_lds_b128) && \
    __has_builtin(__builtin_amdgcn_s_wait_asynccnt)
#define HAVE_ASYNC_LDS 1
#else
#define HAVE_ASYNC_LDS 0
#endif

// One-time f32 -> f16 conversion of both weight matrices into workspace.
__global__ void convert_weights_kernel(const float* __restrict__ wkv,
                                       const float* __restrict__ wz,
                                       _Float16* __restrict__ wkv16,
                                       _Float16* __restrict__ wz16) {
    int i = blockIdx.x * blockDim.x + threadIdx.x;
    if (i < HD_ * D_) {
        wkv16[i] = (_Float16)wkv[i];
        wz16[i]  = (_Float16)wz[i];
    }
}

// Fused dual-GEMM + windowed softmax + weighted reduction.
// Grid: B * (S/64) blocks of 256 threads (8 wave32 waves).
// Each block: 64 S-rows x full HD=128. Wave j owns HD columns [16j,16j+16)
// and reuses each B fragment across 4 row-tiles (4x WMMA per B load).
__global__ void __launch_bounds__(256, 1)
heavy_compressor_kernel(const float*    __restrict__ h,
                        const _Float16* __restrict__ wkv16,
                        const _Float16* __restrict__ wz16,
                        const float*    __restrict__ bias,
                        float*          __restrict__ out) {
    __shared__ float    sf32[2][ROWS][KCHUNK];   // 2 x 32 KB async staging
    __shared__ _Float16 sf16[2][ROWS][KCHUNK];   // 2 x 16 KB converted A

    const int tid   = threadIdx.x;
    const int lane  = tid & 31;
    const int wave  = tid >> 5;
    const int half  = lane >> 4;          // 0: lanes 0-15, 1: lanes 16-31
    const int lrow  = lane & 15;

    const int blocks_per_b = S_ / ROWS;   // 128
    const int bb  = blockIdx.x / blocks_per_b;
    const int blk = blockIdx.x % blocks_per_b;
    const int s0  = blk * ROWS;

    const int colg = wave * 16 + lrow;    // global HD column this lane owns

    v8f cacc[ROWT];                       // c = h @ w_kv^T  (4x 16x16 tiles)
    v8f zacc[ROWT];                       // z = h @ w_z^T
    #pragma unroll
    for (int rt = 0; rt < ROWT; ++rt) { cacc[rt] = (v8f){}; zacc[rt] = (v8f){}; }

    const size_t hbase = ((size_t)bb * S_ + s0) * D_;

    // issue one 64x128 f32 chunk of h into LDS buffer `buf` (8 ops/wave)
    auto issue_chunk = [&](int k0, int buf) {
        #pragma unroll
        for (int it = 0; it < 8; ++it) {
            int chunk = tid + it * 256;       // 0..2047 16B granules
            int row   = chunk >> 5;           // 32 granules per 128-f32 row
            int c4    = (chunk & 31) << 2;
            const float* gsrc = h + hbase + (size_t)row * D_ + k0 + c4;
            float* ldst = &sf32[buf][row][c4];
#if HAVE_ASYNC_LDS
            __builtin_amdgcn_global_load_async_to_lds_b128(
                (v4i*)gsrc, (v4i*)ldst, 0, 0);
#else
            *(v4f*)ldst = *(const v4f*)gsrc;
#endif
        }
    };

    const int NK = D_ / KCHUNK;           // 16
    issue_chunk(0, 0);

    for (int k = 0; k < NK; ++k) {
        const int cur = k & 1;
        const int k0  = k * KCHUNK;

        // software pipeline: issue chunk k+1 while we consume chunk k
        if (k + 1 < NK) {
            issue_chunk(k0 + KCHUNK, 1 - cur);
#if HAVE_ASYNC_LDS
            __builtin_amdgcn_s_wait_asynccnt(8);   // chunk k landed
#endif
        } else {
#if HAVE_ASYNC_LDS
            __builtin_amdgcn_s_wait_asynccnt(0);
#endif
        }
        __syncthreads();

        // cooperative f32 -> f16 convert, once per block (not per wave)
        #pragma unroll
        for (int it = 0; it < 8; ++it) {
            int chunk = tid + it * 256;
            int row   = chunk >> 5;
            int c4    = (chunk & 31) << 2;
            v4f x = *(const v4f*)&sf32[cur][row][c4];
            v4h y;
            y[0] = (_Float16)x.x; y[1] = (_Float16)x.y;
            y[2] = (_Float16)x.z; y[3] = (_Float16)x.w;
            *(v4h*)&sf16[cur][row][c4] = y;
        }
        __syncthreads();

        // ---- 4 WMMA K-steps of 32; each B fragment reused 4x ----
        #pragma unroll
        for (int kk = 0; kk < KCHUNK; kk += 32) {
            // B fragments (32x16 f16): 16 contiguous K halves per lane
            const size_t wofs = (size_t)colg * D_ + k0 + kk + half * 16;
            const v16h bkv = *(const v16h*)(wkv16 + wofs);
            const v16h bz  = *(const v16h*)(wz16  + wofs);

            #pragma unroll
            for (int rt = 0; rt < ROWT; ++rt) {
                // A fragment (16x32 f16): two 8-half K runs from LDS
                const _Float16* ap = &sf16[cur][rt * 16 + lrow][kk + half * 16];
                v8h lo = *(const v8h*)(ap);
                v8h hi = *(const v8h*)(ap + 8);
                v16h a = __builtin_shufflevector(lo, hi,
                    0, 1, 2, 3, 4, 5, 6, 7, 8, 9, 10, 11, 12, 13, 14, 15);

                cacc[rt] = __builtin_amdgcn_wmma_f32_16x16x32_f16(
                    false, a, false, bkv, (short)0, cacc[rt], false, false);
                zacc[rt] = __builtin_amdgcn_wmma_f32_16x16x32_f16(
                    false, a, false, bz,  (short)0, zacc[rt], false, false);
            }
        }
        __syncthreads();   // buffer `cur` free for reuse at k+2
    }

    // ---- epilogue: softmax over m=8 window, entirely within one lane ----
    // C/D layout: VGPR r of lane -> row (half*8 + r), column colg.
    float bv[M_];
    #pragma unroll
    for (int i = 0; i < M_; ++i) bv[i] = bias[i * HD_ + colg];

    #pragma unroll
    for (int rt = 0; rt < ROWT; ++rt) {
        float zb[M_];
        float mx = -3.402823e38f;
        #pragma unroll
        for (int i = 0; i < M_; ++i) {
            zb[i] = zacc[rt][i] + bv[i];
            mx = fmaxf(mx, zb[i]);
        }
        float sum = 0.f, acc = 0.f;
        #pragma unroll
        for (int i = 0; i < M_; ++i) {
            float e = __expf(zb[i] - mx);
            sum += e;
            acc += e * cacc[rt][i];
        }
        const float outv = acc / sum;

        const int nidx = (blk * ROWT + rt) * 2 + half;   // m=8 window index
        const size_t o = ((size_t)bb * N_ + nidx) * HD_ + colg;
        out[o] = outv;                          // tuple element 0
        out[o + (size_t)B_ * N_ * HD_] = outv;  // tuple element 1 (duplicate)
    }
}

extern "C" void kernel_launch(void* const* d_in, const int* in_sizes, int n_in,
                              void* d_out, int out_size, void* d_ws, size_t ws_size,
                              hipStream_t stream) {
    const float* h    = (const float*)d_in[0];
    const float* wkv  = (const float*)d_in[1];
    const float* wz   = (const float*)d_in[2];
    const float* bias = (const float*)d_in[3];
    float* out = (float*)d_out;

    _Float16* wkv16 = (_Float16*)d_ws;
    _Float16* wz16  = wkv16 + (size_t)HD_ * D_;

    convert_weights_kernel<<<(HD_ * D_ + 255) / 256, 256, 0, stream>>>(
        wkv, wz, wkv16, wz16);

    heavy_compressor_kernel<<<B_ * (S_ / ROWS), 256, 0, stream>>>(
        h, wkv16, wz16, bias, out);
}
